// PositionwisePairFeedForward_35914516529609
// MI455X (gfx1250) — matrix operand order repaired and checked by reading, not compiled, not run on hardware
//
#include <hip/hip_runtime.h>
#include <hip/hip_bf16.h>

typedef __attribute__((ext_vector_type(16))) _Float16 v16h;
typedef __attribute__((ext_vector_type(8)))  _Float16 v8h;
typedef __attribute__((ext_vector_type(8)))  float    v8f;

#define WMMA_F16(a, b, c) \
  __builtin_amdgcn_wmma_f32_16x16x32_f16(false, (a), false, (b), (short)0, (c), false, false)

// ---------------- fragment loaders ----------------
// A-frag 16x32 f16 (MxK): lane L: row = L&15; K = {8*hi..8*hi+7} U {16+8*hi..16+8*hi+7}, hi = L>>4
static __device__ __forceinline__ v16h load_a_f16(const _Float16* src, int ld, int row0, int k0, int lane) {
  const int r = lane & 15, hi = lane >> 4;
  const _Float16* p = src + (size_t)(row0 + r) * ld + k0 + hi * 8;
  v8h x0 = *(const v8h*)p;
  v8h x1 = *(const v8h*)(p + 16);
  v16h a;
#pragma unroll
  for (int j = 0; j < 8; ++j) { a[j] = x0[j]; a[8 + j] = x1[j]; }
  return a;
}

static __device__ __forceinline__ v16h load_a_f32(const float* src, int ld, int row0, int k0, int lane) {
  const int r = lane & 15, hi = lane >> 4;
  const float* p = src + (size_t)(row0 + r) * ld + k0 + hi * 8;
  v16h a;
#pragma unroll
  for (int j = 0; j < 8; ++j) { a[j] = (_Float16)p[j]; a[8 + j] = (_Float16)p[16 + j]; }
  return a;
}

// B-frag 32x16 (KxN) built from a ROW-MAJOR matrix R where B[k][n] = R[n0+n][k0+k]
// lane L: column = L&15; K = {8*hi..8*hi+7} U {16+8*hi..16+8*hi+7}, hi = L>>4
static __device__ __forceinline__ v16h load_b_f16(const _Float16* src, int ld, int n0, int k0, int lane) {
  const int c = lane & 15, hi = lane >> 4;
  const _Float16* p = src + (size_t)(n0 + c) * ld + k0 + hi * 8;
  v8h x0 = *(const v8h*)p;
  v8h x1 = *(const v8h*)(p + 16);
  v16h b;
#pragma unroll
  for (int j = 0; j < 8; ++j) { b[j] = x0[j]; b[8 + j] = x1[j]; }
  return b;
}

static __device__ __forceinline__ v16h load_b_f32(const float* src, int ld, int n0, int k0, int lane) {
  const int c = lane & 15, hi = lane >> 4;
  const float* p = src + (size_t)(n0 + c) * ld + k0 + hi * 8;
  v16h b;
#pragma unroll
  for (int j = 0; j < 8; ++j) { b[j] = (_Float16)p[j]; b[8 + j] = (_Float16)p[16 + j]; }
  return b;
}

// reductions within each 16-lane half (rows of a C-fragment live across 16 lanes)
static __device__ __forceinline__ float half_max(float v) {
#pragma unroll
  for (int m = 1; m < 16; m <<= 1) v = fmaxf(v, __shfl_xor(v, m, 32));
  return v;
}
static __device__ __forceinline__ float half_sum(float v) {
#pragma unroll
  for (int m = 1; m < 16; m <<= 1) v += __shfl_xor(v, m, 32);
  return v;
}

// ---------------- kernel 1: transpose + f32->f16 convert: At[256][8192] = A[8192][256]^T ----
__global__ __launch_bounds__(256) void transpose_cvt_kernel(const float* __restrict__ A,
                                                            _Float16* __restrict__ At) {
  __shared__ float tile[32][33];
  const int bx = blockIdx.x * 32;  // column of A (0..255)
  const int by = blockIdx.y * 32;  // row of A (0..8191)
  const int tx = threadIdx.x, ty = threadIdx.y;
#pragma unroll
  for (int j = 0; j < 32; j += 8)
    tile[ty + j][tx] = A[(size_t)(by + ty + j) * 256 + bx + tx];
  __syncthreads();
#pragma unroll
  for (int j = 0; j < 32; j += 8)
    At[(size_t)(bx + ty + j) * 8192 + by + tx] = (_Float16)tile[tx][ty + j];
}

// ---------------- kernel 2: projection  out_f16[8192][256] = X[8192][256] * W^T + bias -------
__global__ __launch_bounds__(64) void proj_kernel(const float* __restrict__ X,
                                                  const float* __restrict__ W,
                                                  const float* __restrict__ bias,
                                                  _Float16* __restrict__ outh) {
  const int lane = threadIdx.x & 31, wave = threadIdx.x >> 5;
  const int m0 = (blockIdx.x * 2 + wave) * 16;
  const int hi = lane >> 4, c = lane & 15;
  v16h a[8];
#pragma unroll
  for (int kc = 0; kc < 8; ++kc) a[kc] = load_a_f32(X, 256, m0, kc * 32, lane);
#pragma unroll
  for (int dt = 0; dt < 16; ++dt) {
    v8f acc = {};
#pragma unroll
    for (int kc = 0; kc < 8; ++kc) {
      v16h b = load_b_f32(W, 256, dt * 16, kc * 32, lane);
      acc = WMMA_F16(a[kc], b, acc);
    }
    const float bb = bias[dt * 16 + c];
#pragma unroll
    for (int i = 0; i < 8; ++i)
      outh[(size_t)(m0 + i + 8 * hi) * 256 + dt * 16 + c] = (_Float16)(acc[i] + bb);
  }
}

// ---------------- kernel 3: fused cross-attention (flash style, async-LDS staged) ------------
// ge2ac[M][256] = softmax(ge * ac^T) * anchor
// Block: 128 threads = 4 waves, each wave owns a 16-row Q tile (64 rows / block).
// Per 64-column step, the block async-copies the shared K tile (64x256 f16, contiguous 32KB)
// and V tile (anchorT rows, 256x64 f16, 32KB) into double-buffered LDS.
__global__ __launch_bounds__(128) void flash_kernel(const _Float16* __restrict__ geh,
                                                    const _Float16* __restrict__ ach,
                                                    const _Float16* __restrict__ anchorT,
                                                    float* __restrict__ ge2ac) {
  __shared__ _Float16 ktile[2][64 * 256];   // 2 x 32 KB
  __shared__ _Float16 vtile[2][256 * 64];   // 2 x 32 KB
  __shared__ _Float16 pstage[4][16 * 64];   // 4 x 2 KB  P staging (C-layout -> A-layout)
  const int tid = threadIdx.x;
  const int lane = tid & 31, wave = tid >> 5;
  const int m0 = (blockIdx.x * 4 + wave) * 16;
  const int hi = lane >> 4, c = lane & 15;
  _Float16* myp = pstage[wave];

  const unsigned k_lds0 = (unsigned)(uintptr_t)&ktile[0][0];
  const unsigned k_lds1 = (unsigned)(uintptr_t)&ktile[1][0];
  const unsigned v_lds0 = (unsigned)(uintptr_t)&vtile[0][0];
  const unsigned v_lds1 = (unsigned)(uintptr_t)&vtile[1][0];
  const unsigned long long ach_a = (unsigned long long)(uintptr_t)ach;
  const unsigned long long anT_a = (unsigned long long)(uintptr_t)anchorT;

  // issue one 64-column stage: 4096 b128 chunks, 32 per thread, 32 async instr per wave
  auto issue_stage = [&](int n0, int buf) {
    const unsigned kb = buf ? k_lds1 : k_lds0;
    const unsigned vb = buf ? v_lds1 : v_lds0;
#pragma unroll
    for (int i = 0; i < 32; ++i) {
      const int cid = i * 128 + tid;           // i<16 -> K tile, i>=16 -> V tile (uniform)
      unsigned long long gaddr;
      unsigned ldsa;
      if (cid < 2048) {                        // K tile: contiguous 32KB at ach + n0*512B
        gaddr = ach_a + (size_t)n0 * 512 + (size_t)cid * 16;
        ldsa  = kb + (unsigned)cid * 16;
      } else {                                 // V tile: 256 rows x 128B from anchorT
        const int v = cid - 2048, row = v >> 3, ch = v & 7;
        gaddr = anT_a + (size_t)row * 16384 + (size_t)n0 * 2 + (size_t)ch * 16;
        ldsa  = vb + (unsigned)(row * 128 + ch * 16);
      }
      asm volatile("global_load_async_to_lds_b128 %0, %1, off"
                   :: "v"(ldsa), "v"(gaddr) : "memory");
    }
  };

  v16h q[8];
#pragma unroll
  for (int kc = 0; kc < 8; ++kc) q[kc] = load_a_f16(geh, 256, m0, kc * 32, lane);

  v8f acc[16];
#pragma unroll
  for (int dt = 0; dt < 16; ++dt) acc[dt] = (v8f){};
  float mr[8], lr[8];
#pragma unroll
  for (int i = 0; i < 8; ++i) { mr[i] = -3.0e38f; lr[i] = 0.f; }

  issue_stage(0, 0);
  for (int it = 0; it < 128; ++it) {
    const int buf = it & 1;
    if (it + 1 < 128) {
      issue_stage((it + 1) * 64, buf ^ 1);
      asm volatile("s_wait_asynccnt 32" ::: "memory");  // in-order: current stage resident
    } else {
      asm volatile("s_wait_asynccnt 0" ::: "memory");
    }
    __syncthreads();
    const _Float16* K = ktile[buf];
    const _Float16* V = vtile[buf];

    // S tile: 16 x 64
    v8f s[4];
#pragma unroll
    for (int t = 0; t < 4; ++t) s[t] = (v8f){};
#pragma unroll
    for (int kc = 0; kc < 8; ++kc) {
#pragma unroll
      for (int t = 0; t < 4; ++t) {
        v16h b = load_b_f16(K, 256, t * 16, kc * 32, lane);
        s[t] = WMMA_F16(q[kc], b, s[t]);
      }
    }

    // online softmax (row = i + 8*hi, col = c within each half-wave)
#pragma unroll
    for (int i = 0; i < 8; ++i) {
      float vm = fmaxf(fmaxf(s[0][i], s[1][i]), fmaxf(s[2][i], s[3][i]));
      vm = half_max(vm);
      const float mnew = fmaxf(mr[i], vm);
      const float scale = __expf(mr[i] - mnew);
      mr[i] = mnew;
      float ps = 0.f;
#pragma unroll
      for (int t = 0; t < 4; ++t) {
        const float p = __expf(s[t][i] - mnew);
        ps += p;
        myp[(i + 8 * hi) * 64 + t * 16 + c] = (_Float16)p;
      }
      lr[i] = lr[i] * scale + half_sum(ps);
#pragma unroll
      for (int dt = 0; dt < 16; ++dt) acc[dt][i] *= scale;
    }
    asm volatile("s_wait_dscnt 0" ::: "memory");
    v16h pf0 = load_a_f16(myp, 64, 0, 0, lane);   // P as two 16x32 A-frags
    v16h pf1 = load_a_f16(myp, 64, 0, 32, lane);
    asm volatile("" ::: "memory");
#pragma unroll
    for (int dt = 0; dt < 16; ++dt) {
      acc[dt] = WMMA_F16(pf0, load_b_f16(V, 64, dt * 16, 0, lane), acc[dt]);
      acc[dt] = WMMA_F16(pf1, load_b_f16(V, 64, dt * 16, 32, lane), acc[dt]);
    }
    __syncthreads();   // all waves done reading buf before it is refilled
  }

  // normalize + store
#pragma unroll
  for (int i = 0; i < 8; ++i) {
    const float inv = 1.0f / lr[i];
#pragma unroll
    for (int dt = 0; dt < 16; ++dt)
      ge2ac[(size_t)(m0 + i + 8 * hi) * 256 + dt * 16 + c] = acc[dt][i] * inv;
  }
}

// ---------------- kernel 4: MLP head ---------------------------------------------------------
static __device__ __forceinline__ float dis_elem(const float* g, const float* e, int row, int k) {
  if (k < 256) return g[(size_t)row * 256 + k];
  if (k < 512) return e[(size_t)row * 256 + (k - 256)];
  return g[(size_t)row * 256 + (k - 512)] - e[(size_t)row * 256 + (k - 512)];
}
static __device__ __forceinline__ v16h load_a_dis(const float* g, const float* e, int m0, int k0, int lane) {
  const int r = lane & 15, hi = lane >> 4;
  const int row = m0 + r, kb = k0 + 8 * hi;
  v16h a;
#pragma unroll
  for (int j = 0; j < 8; ++j) {
    a[j]     = (_Float16)dis_elem(g, e, row, kb + j);
    a[8 + j] = (_Float16)dis_elem(g, e, row, kb + 16 + j);
  }
  return a;
}

__global__ __launch_bounds__(128) void mlp_kernel(const float* __restrict__ guess,
                                                  const float* __restrict__ ge2ac,
                                                  const float* __restrict__ W1, const float* __restrict__ b1,
                                                  const float* __restrict__ a1p,
                                                  const float* __restrict__ W2, const float* __restrict__ b2,
                                                  const float* __restrict__ a2p,
                                                  const float* __restrict__ W3, const float* __restrict__ b3p,
                                                  float* __restrict__ out) {
  __shared__ _Float16 h1buf[4][16 * 256];
  __shared__ _Float16 h2buf[4][16 * 128];
  const int lane = threadIdx.x & 31, wave = threadIdx.x >> 5;
  const int m0 = (blockIdx.x * 4 + wave) * 16;
  const int hi = lane >> 4, c = lane & 15;
  const float a1 = a1p[0], a2 = a2p[0];
  _Float16* h1 = h1buf[wave];
  _Float16* h2 = h2buf[wave];

  // stage 1: h1 = prelu(dis_x * W1^T + b1), dis_x synthesized on the fly (K = 768)
  v16h a[24];
#pragma unroll
  for (int kc = 0; kc < 24; ++kc) a[kc] = load_a_dis(guess, ge2ac, m0, kc * 32, lane);
#pragma unroll
  for (int dt = 0; dt < 16; ++dt) {
    v8f acc = {};
#pragma unroll
    for (int kc = 0; kc < 24; ++kc) {
      v16h b = load_b_f32(W1, 768, dt * 16, kc * 32, lane);
      acc = WMMA_F16(a[kc], b, acc);
    }
    const float bb = b1[dt * 16 + c];
#pragma unroll
    for (int i = 0; i < 8; ++i) {
      float x = acc[i] + bb;
      x = x >= 0.f ? x : a1 * x;
      h1[(i + 8 * hi) * 256 + dt * 16 + c] = (_Float16)x;
    }
  }
  asm volatile("s_wait_dscnt 0" ::: "memory");

  // stage 2: h2 = prelu(h1 * W2^T + b2)  (K = 256, 128 outputs)
  v16h ah[8];
#pragma unroll
  for (int kc = 0; kc < 8; ++kc) ah[kc] = load_a_f16(h1, 256, 0, kc * 32, lane);
#pragma unroll
  for (int dt = 0; dt < 8; ++dt) {
    v8f acc = {};
#pragma unroll
    for (int kc = 0; kc < 8; ++kc) {
      v16h b = load_b_f32(W2, 256, dt * 16, kc * 32, lane);
      acc = WMMA_F16(ah[kc], b, acc);
    }
    const float bb = b2[dt * 16 + c];
#pragma unroll
    for (int i = 0; i < 8; ++i) {
      float x = acc[i] + bb;
      x = x >= 0.f ? x : a2 * x;
      h2[(i + 8 * hi) * 128 + dt * 16 + c] = (_Float16)x;
    }
  }
  asm volatile("s_wait_dscnt 0" ::: "memory");

  // stage 3: out = h2 * W3^T + b3  (one scalar per row)
  if (lane < 16) {
    float s = b3p[0];
#pragma unroll 8
    for (int k = 0; k < 128; ++k) s += (float)h2[lane * 128 + k] * W3[k];
    out[m0 + lane] = s;
  }
}

// ---------------- launcher -------------------------------------------------------------------
extern "C" void kernel_launch(void* const* d_in, const int* in_sizes, int n_in,
                              void* d_out, int out_size, void* d_ws, size_t ws_size,
                              hipStream_t stream) {
  const float* anchor = (const float*)d_in[0];
  const float* guess  = (const float*)d_in[1];
  const float* Wa = (const float*)d_in[2];
  const float* ba = (const float*)d_in[3];
  const float* Wg = (const float*)d_in[4];
  const float* bg = (const float*)d_in[5];
  const float* W1 = (const float*)d_in[6];
  const float* b1 = (const float*)d_in[7];
  const float* a1 = (const float*)d_in[8];
  const float* W2 = (const float*)d_in[9];
  const float* b2 = (const float*)d_in[10];
  const float* a2 = (const float*)d_in[11];
  const float* W3 = (const float*)d_in[12];
  const float* b3 = (const float*)d_in[13];

  char* ws = (char*)d_ws;
  _Float16* ach     = (_Float16*)(ws);                       // 4 MB: ac  f16 [8192][256]
  _Float16* geh     = (_Float16*)(ws + ((size_t)4 << 20));   // 4 MB: ge  f16 [8192][256]
  _Float16* anchorT = (_Float16*)(ws + ((size_t)8 << 20));   // 4 MB: anchor^T f16 [256][8192]
  float*    ge2ac   = (float*)   (ws + ((size_t)12 << 20));  // 8 MB: ge2ac f32 [8192][256]

  transpose_cvt_kernel<<<dim3(8, 256), dim3(32, 8), 0, stream>>>(anchor, anchorT);
  proj_kernel<<<256, 64, 0, stream>>>(anchor, Wa, ba, ach);
  proj_kernel<<<256, 64, 0, stream>>>(guess,  Wg, bg, geh);
  flash_kernel<<<128, 128, 0, stream>>>(geh, ach, anchorT, ge2ac);
  mlp_kernel<<<128, 128, 0, stream>>>(guess, ge2ac, W1, b1, a1, W2, b2, a2, W3, b3, (float*)d_out);
}
